// ImageMLA_69836168233756
// MI455X (gfx1250) — compile-verified, matrix operand order checked by reading, not run on hardware
//
#include <hip/hip_runtime.h>
#include <hip/hip_bf16.h>
#include <math.h>

typedef __attribute__((ext_vector_type(16))) _Float16 v16h;
typedef __attribute__((ext_vector_type(8)))  float    v8f;
typedef unsigned int v4u __attribute__((ext_vector_type(4)));
typedef int          v8i __attribute__((ext_vector_type(8)));
typedef int          v4i __attribute__((ext_vector_type(4)));

#define DIMC   128
#define HEADS  4
#define SDIM   4096
#define QRANK  32
#define KVRANK 64
#define ROPE   64
#define NOPE   64
#define VD     64
#define HD     128     /* NOPE+ROPE */
#define OCH    256     /* HEADS*VD  */
#define BATCH  2
#define EPSF   1e-5f
#define LN1E4_16 0.5756462732485115f  /* ln(10000)/16 */

// ---------------------------------------------------------------------------
// CDNA5 WMMA 16-bit fragment loader (A M-major / B from N-major storage).
//   row = lane & 15 ; VGPR v holds halves {k,k+1},
//   k = (v>=4 ? 16:0) + (lane>>4)*8 + (v&3)*2
// ---------------------------------------------------------------------------
__device__ __forceinline__ v16h ld_frag(const _Float16* p, int ld, int row0, int col0) {
  const int lane = threadIdx.x & 31;
  const int r    = lane & 15;
  const int khi  = (lane >> 4) << 3;
  const _Float16* q = p + (size_t)(row0 + r) * ld + col0;
  v16h out;
#pragma unroll
  for (int v = 0; v < 8; ++v) {
    int k = ((v & 4) << 2) + khi + ((v & 3) << 1);
    out[2 * v]     = q[k];
    out[2 * v + 1] = q[k + 1];
  }
  return out;
}

// ---------------------------------------------------------------------------
// 16-lane (DPP16 row) reductions via rotate butterflies: row_ror:1/2/4/8.
// VALU-only: co-executes with matrix-pipe WMMAs, leaves the LDS pipe free.
// DPP control must be an ICE -> template parameter.
// ---------------------------------------------------------------------------
template <int CTRL>
__device__ __forceinline__ float dpp_mov(float v) {
  return __int_as_float(
      __builtin_amdgcn_update_dpp(0, __float_as_int(v), CTRL, 0xf, 0xf, true));
}
__device__ __forceinline__ float row16_max(float v) {
  v = fmaxf(v, dpp_mov<0x121>(v));   // row_ror:1
  v = fmaxf(v, dpp_mov<0x122>(v));   // row_ror:2
  v = fmaxf(v, dpp_mov<0x124>(v));   // row_ror:4
  v = fmaxf(v, dpp_mov<0x128>(v));   // row_ror:8
  return v;
}
__device__ __forceinline__ float row16_sum(float v) {
  v += dpp_mov<0x121>(v);
  v += dpp_mov<0x122>(v);
  v += dpp_mov<0x124>(v);
  v += dpp_mov<0x128>(v);
  return v;
}

// ---------------------------------------------------------------------------
// Tensor Data Mover: async DMA of a (up to 2-D) tile from global into LDS.
// D# per CDNA5 ISA ch.8: group0 = {ctrl, lds_addr, global_addr, type=2},
// group1 = {data_size, tensor dims, tile dims, dim0 stride}. Higher groups
// zero (2-D tensor). Tracked by TENSORcnt.  (6-arg toolchain variant.)
// ---------------------------------------------------------------------------
__device__ __forceinline__ void tdm_load_2d(unsigned lds_off, const void* gptr,
                                            unsigned ds_log2,      /* 0=1B..3=8B */
                                            unsigned tile_d0, unsigned tile_d1,
                                            unsigned tens_d0, unsigned tens_d1,
                                            unsigned stride0 /* elements */) {
  unsigned long long ga = (unsigned long long)gptr;
  v4u g0;
  g0[0] = 1u;                                   // count=1, user descriptor
  g0[1] = lds_off;                              // LDS byte address
  g0[2] = (unsigned)(ga & 0xFFFFFFFFu);         // global_addr[31:0]
  g0[3] = (unsigned)(ga >> 32) | 0x80000000u;   // global_addr[56:32] | type=2
  v8i g1;
  g1[0] = (int)(ds_log2 << 16);                 // workgroup_mask=0, data_size
  g1[1] = (int)((tens_d0 & 0xFFFFu) << 16);     // tensor_dim0[15:0] @ bits63:48
  g1[2] = (int)(((tens_d0 >> 16) & 0xFFFFu) | ((tens_d1 & 0xFFFFu) << 16));
  g1[3] = (int)(((tens_d1 >> 16) & 0xFFFFu) | ((tile_d0 & 0xFFFFu) << 16));
  g1[4] = (int)(tile_d1 & 0xFFFFu);             // tile_dim1 | tile_dim2=0
  g1[5] = (int)stride0;                         // tensor_dim0_stride[31:0]
  g1[6] = 0;                                    // stride0[47:32] | dim1_stride lo
  g1[7] = 0;
  v4i z4 = {0, 0, 0, 0};
  v8i z8 = {0, 0, 0, 0, 0, 0, 0, 0};
  __builtin_amdgcn_tensor_load_to_lds(g0, g1, z4, z4, z8, 0);
}

// ---------------------------------------------------------------------------
// K1: x -> q1 (32ch) + kv1 (128ch); accumulate layernorm stats per batch.
// ---------------------------------------------------------------------------
__global__ __launch_bounds__(256) void proj1_kernel(
    const float* __restrict__ x,
    const float* __restrict__ q1_w, const float* __restrict__ q1_b,
    const float* __restrict__ kv1_w, const float* __restrict__ kv1_b,
    float* __restrict__ q1_ws, float* __restrict__ kv1_ws,
    float* __restrict__ stats1) {
  __shared__ float xsh[DIMC][64];
  const int tile = blockIdx.x;           // B*(S/64) = 128
  const int b = tile >> 6;
  const int s0 = (tile & 63) * 64;
  const int tid = threadIdx.x;
#pragma unroll
  for (int i = 0; i < (DIMC * 64) / 256; ++i) {
    int idx = tid + i * 256;
    int c = idx >> 6, sl = idx & 63;
    xsh[c][sl] = x[((size_t)b * DIMC + c) * SDIM + s0 + sl];
  }
  __syncthreads();
  float qs = 0.f, qss = 0.f, ks = 0.f, kss = 0.f;
  for (int i = 0; i < (160 * 64) / 256; ++i) {   // 40 outputs/thread
    int item = tid + i * 256;
    int co = item >> 6, sl = item & 63;
    if (co < QRANK) {
      float acc = q1_b[co];
      const float* wr = q1_w + (size_t)co * DIMC;
      for (int c = 0; c < DIMC; ++c) acc += wr[c] * xsh[c][sl];
      q1_ws[((size_t)b * QRANK + co) * SDIM + s0 + sl] = acc;
      qs += acc; qss += acc * acc;
    } else {
      int ck = co - QRANK;
      float acc = kv1_b[ck];
      const float* wr = kv1_w + (size_t)ck * DIMC;
      for (int c = 0; c < DIMC; ++c) acc += wr[c] * xsh[c][sl];
      kv1_ws[((size_t)b * DIMC + ck) * SDIM + s0 + sl] = acc;
      ks += acc; kss += acc * acc;
    }
  }
#pragma unroll
  for (int m = 1; m < 32; m <<= 1) {
    qs += __shfl_xor(qs, m, 32);  qss += __shfl_xor(qss, m, 32);
    ks += __shfl_xor(ks, m, 32);  kss += __shfl_xor(kss, m, 32);
  }
  if ((tid & 31) == 0) {
    atomicAdd(&stats1[b * 4 + 0], qs);
    atomicAdd(&stats1[b * 4 + 1], qss);
    atomicAdd(&stats1[b * 4 + 2], ks);
    atomicAdd(&stats1[b * 4 + 3], kss);
  }
}

__global__ void stats1_kernel(const float* __restrict__ stats1, float* __restrict__ norm1) {
  int b = threadIdx.x;
  if (b >= BATCH) return;
  float nq = (float)QRANK * (float)SDIM;
  float nk = (float)DIMC * (float)SDIM;
  float qm = stats1[b * 4 + 0] / nq;
  float qv = stats1[b * 4 + 1] / nq - qm * qm;
  float km = stats1[b * 4 + 2] / nk;
  float kv = stats1[b * 4 + 3] / nk - km * km;
  norm1[b * 4 + 0] = qm; norm1[b * 4 + 1] = rsqrtf(qv + EPSF);
  norm1[b * 4 + 2] = km; norm1[b * 4 + 3] = rsqrtf(kv + EPSF);
}

// ---------------------------------------------------------------------------
// K3: gn(q1)->q2 conv->rope -> Q(f16); gn(kv1)->rope K_rope(f16) + kv_c(f32)
// + stats for the inner (kvn) norm.
// ---------------------------------------------------------------------------
__global__ __launch_bounds__(256) void mid_kernel(
    const float* __restrict__ q1_ws, const float* __restrict__ kv1_ws,
    const float* __restrict__ norm1,
    const float* __restrict__ q_gn_g, const float* __restrict__ q_gn_b,
    const float* __restrict__ q2_w, const float* __restrict__ q2_b,
    const float* __restrict__ kv_gn_g, const float* __restrict__ kv_gn_b,
    _Float16* __restrict__ Qh, _Float16* __restrict__ Krope,
    float* __restrict__ kvc_ws, float* __restrict__ stats2) {
  __shared__ float q1sh[QRANK][64];
  const int tile = blockIdx.x;
  const int b = tile >> 6;
  const int s0 = (tile & 63) * 64;
  const int tid = threadIdx.x;
  const float qm = norm1[b * 4 + 0], qi = norm1[b * 4 + 1];
  const float km = norm1[b * 4 + 2], ki = norm1[b * 4 + 3];
#pragma unroll
  for (int i = 0; i < (QRANK * 64) / 256; ++i) {
    int idx = tid + i * 256;
    int c = idx >> 6, sl = idx & 63;
    q1sh[c][sl] = (q1_ws[((size_t)b * QRANK + c) * SDIM + s0 + sl] - qm) * qi * q_gn_g[c] + q_gn_b[c];
  }
  __syncthreads();
  // q2 conv (channel pairs) + rope
  for (int i = 0; i < 64; ++i) {
    int item = tid + i * 256;
    int pr = item >> 6, sl = item & 63;
    int ch0 = pr * 2;
    float a0 = q2_b[ch0], a1 = q2_b[ch0 + 1];
    const float* w0 = q2_w + (size_t)ch0 * QRANK;
    const float* w1 = w0 + QRANK;
    for (int c = 0; c < QRANK; ++c) {
      float qv = q1sh[c][sl];
      a0 += w0[c] * qv; a1 += w1[c] * qv;
    }
    int h = ch0 >> 7, d = ch0 & 127;
    int s = s0 + sl;
    if (d >= NOPE) {
      int j = (d - NOPE) >> 1;
      float inv = __expf(-(float)(j >> 1) * LN1E4_16);
      float coord = (float)((j & 1) ? (s & 63) : (s >> 6));
      float fr = __sinf(coord * inv), fi = __cosf(coord * inv);
      float rr = a0 * fr - a1 * fi;
      float ri = a0 * fi + a1 * fr;
      a0 = rr; a1 = ri;
    }
    _Float16* qp = Qh + ((size_t)(b * HEADS + h) * SDIM + s) * HD + d;
    qp[0] = (_Float16)a0; qp[1] = (_Float16)a1;
  }
  // kv split: rope half -> Krope f16; content half -> kvc fp32 + stats
  float ss = 0.f, sss = 0.f;
  for (int i = 0; i < 16; ++i) {
    int item = tid + i * 256;
    int pr = item >> 6, sl = item & 63;
    int ch0 = pr * 2;
    int s = s0 + sl;
    float v0 = (kv1_ws[((size_t)b * DIMC + ch0) * SDIM + s] - km) * ki * kv_gn_g[ch0] + kv_gn_b[ch0];
    float v1 = (kv1_ws[((size_t)b * DIMC + ch0 + 1) * SDIM + s] - km) * ki * kv_gn_g[ch0 + 1] + kv_gn_b[ch0 + 1];
    if (ch0 < ROPE) {
      int j = pr;
      float inv = __expf(-(float)(j >> 1) * LN1E4_16);
      float coord = (float)((j & 1) ? (s & 63) : (s >> 6));
      float fr = __sinf(coord * inv), fi = __cosf(coord * inv);
      float rr = v0 * fr - v1 * fi;
      float ri = v0 * fi + v1 * fr;
      _Float16* kp = Krope + ((size_t)b * SDIM + s) * ROPE + ch0;
      kp[0] = (_Float16)rr; kp[1] = (_Float16)ri;
    } else {
      int cc = ch0 - ROPE;
      kvc_ws[((size_t)b * KVRANK + cc) * SDIM + s] = v0;
      kvc_ws[((size_t)b * KVRANK + cc + 1) * SDIM + s] = v1;
      ss += v0 + v1; sss += v0 * v0 + v1 * v1;
    }
  }
#pragma unroll
  for (int m = 1; m < 32; m <<= 1) {
    ss += __shfl_xor(ss, m, 32);
    sss += __shfl_xor(sss, m, 32);
  }
  if ((tid & 31) == 0) {
    atomicAdd(&stats2[b * 2 + 0], ss);
    atomicAdd(&stats2[b * 2 + 1], sss);
  }
}

__global__ void stats2_kernel(const float* __restrict__ stats2, float* __restrict__ norm2) {
  int b = threadIdx.x;
  if (b >= BATCH) return;
  float n = (float)KVRANK * (float)SDIM;
  float m = stats2[b * 2 + 0] / n;
  float v = stats2[b * 2 + 1] / n - m * m;
  norm2[b * 2 + 0] = m;
  norm2[b * 2 + 1] = rsqrtf(v + EPSF);
}

// ---------------------------------------------------------------------------
// K5: gn(kv_c) -> kvup conv -> K_nope (b,h,s,128) / V^T (b,h,d,s) in f16,
// broadcast K_rope into K.
// ---------------------------------------------------------------------------
__global__ __launch_bounds__(256) void kvup_kernel(
    const float* __restrict__ kvc_ws, const float* __restrict__ norm2,
    const float* __restrict__ kvn_g, const float* __restrict__ kvn_b,
    const float* __restrict__ kvup_w, const float* __restrict__ kvup_b,
    const _Float16* __restrict__ Krope,
    _Float16* __restrict__ Kh, _Float16* __restrict__ VhT) {
  __shared__ float kvsh[KVRANK][64];
  const int tile = blockIdx.x;
  const int b = tile >> 6;
  const int s0 = (tile & 63) * 64;
  const int tid = threadIdx.x;
  const float m2 = norm2[b * 2 + 0], i2 = norm2[b * 2 + 1];
#pragma unroll
  for (int i = 0; i < (KVRANK * 64) / 256; ++i) {
    int idx = tid + i * 256;
    int c = idx >> 6, sl = idx & 63;
    kvsh[c][sl] = (kvc_ws[((size_t)b * KVRANK + c) * SDIM + s0 + sl] - m2) * i2 * kvn_g[c] + kvn_b[c];
  }
  __syncthreads();
  for (int i = 0; i < 128; ++i) {
    int item = tid + i * 256;
    int oc = item >> 6, sl = item & 63;
    float acc = kvup_b[oc];
    const float* wr = kvup_w + (size_t)oc * KVRANK;
    for (int c = 0; c < KVRANK; ++c) acc += wr[c] * kvsh[c][sl];
    int h = oc >> 7, d = oc & 127;
    int s = s0 + sl;
    if (d < NOPE)
      Kh[((size_t)(b * HEADS + h) * SDIM + s) * HD + d] = (_Float16)acc;
    else  // V transposed: [b][h][d][s]
      VhT[((size_t)(b * HEADS + h) * VD + (d - NOPE)) * SDIM + s] = (_Float16)acc;
  }
  for (int i = 0; i < 64; ++i) {   // broadcast rope into all heads
    int item = tid + i * 256;
    int hd = item >> 6, sl = item & 63;
    int h = hd >> 6, dr = hd & 63;
    int s = s0 + sl;
    Kh[((size_t)(b * HEADS + h) * SDIM + s) * HD + NOPE + dr] =
        Krope[((size_t)b * SDIM + s) * ROPE + dr];
  }
}

// ---------------------------------------------------------------------------
// K6: flash attention, TDM double-buffered K/V staging.
// Dynamic LDS layout (byte offsets):
//   [    0,  8192)  K buf0   (32 x 128 f16)
//   [ 8192, 16384)  K buf1
//   [16384, 20480)  Vt buf0  (64 x 32 f16)
//   [20480, 24576)  Vt buf1
//   [24576, 28672)  P        (4 waves x 16 x 32 f16)
// ---------------------------------------------------------------------------
#define ATTN_LDS_BYTES 28672

__global__ __launch_bounds__(128) void attn_kernel(
    const _Float16* __restrict__ Qh, const _Float16* __restrict__ Kh,
    const _Float16* __restrict__ VhT, float* __restrict__ o_ws) {
  extern __shared__ __align__(16) char smem[];
  _Float16* KshBuf = (_Float16*)smem;               // 2 x 4096 halves
  _Float16* VtBuf  = (_Float16*)(smem + 16384);     // 2 x 2048 halves
  _Float16* PshAll = (_Float16*)(smem + 24576);     // 4 x 512 halves

  const int blk = blockIdx.x;            // B*HEADS*(S/64) = 512
  const int qt64 = blk & 63;
  const int h = (blk >> 6) & 3;
  const int b = blk >> 8;
  const int tid = threadIdx.x;
  const int wave = tid >> 5;
  const int lane = tid & 31;
  const int lanehi = lane >> 4;
  const int qbase = qt64 * 64 + wave * 16;

  const _Float16* Qg  = Qh  + (size_t)(b * HEADS + h) * SDIM * HD;
  const _Float16* Kg  = Kh  + (size_t)(b * HEADS + h) * SDIM * HD;
  const _Float16* VgT = VhT + (size_t)(b * HEADS + h) * VD * SDIM;

  const float scale = rsqrtf((float)HD);

  // Q fragments, pre-scaled by softmax scale (folds 16 v_mul per chunk away)
  v16h qa[4];
#pragma unroll
  for (int kk = 0; kk < 4; ++kk) {
    qa[kk] = ld_frag(Qg + (size_t)qbase * HD, HD, 0, kk * 32);
#pragma unroll
    for (int e = 0; e < 16; ++e)
      qa[kk][e] = (_Float16)((float)qa[kk][e] * scale);
  }

  v8f oc[4];
  float mrow[8], lrow[8];
#pragma unroll
  for (int d = 0; d < 4; ++d)
#pragma unroll
    for (int e = 0; e < 8; ++e) oc[d][e] = 0.f;
#pragma unroll
  for (int v = 0; v < 8; ++v) { mrow[v] = -INFINITY; lrow[v] = 0.f; }

  const int NCH = SDIM / 32;

  // prologue: DMA chunk 0 into buffer 0
  if (wave == 0) {
    tdm_load_2d(0u, Kg, 3, 1024, 1, 1024, 1, 1024);                 // 8 KB, 1-D
    tdm_load_2d(16384u, VgT, 1, 32, VD, SDIM, VD, SDIM);            // 64 rows x 32
    __builtin_amdgcn_s_wait_tensorcnt((unsigned short)0);
  }
  __syncthreads();

  for (int chunk = 0; chunk < NCH; ++chunk) {
    const int cur = chunk & 1;
    // kick off DMA for the next chunk into the other buffer
    if (wave == 0 && chunk + 1 < NCH) {
      const int kb2 = (chunk + 1) * 32;
      tdm_load_2d(cur ? 0u : 8192u, Kg + (size_t)kb2 * HD,
                  3, 1024, 1, 1024, 1, 1024);
      tdm_load_2d(cur ? 16384u : 20480u, VgT + kb2,
                  1, 32, VD, SDIM, VD, SDIM);
    }
    const _Float16* Ksh = KshBuf + cur * 4096;
    const _Float16* Vt  = VtBuf + cur * 2048;

    // scores: Q(16x128) x K_chunk^T -> two 16x16 f32 tiles.
    // B-fragment loads software-pipelined one WMMA ahead so the DS waits
    // overlap with matrix-pipe execution.
    v8f s0, s1;
#pragma unroll
    for (int e = 0; e < 8; ++e) { s0[e] = 0.f; s1[e] = 0.f; }
    v16h kb0 = ld_frag(Ksh, HD, 0, 0);
    v16h kb1 = ld_frag(Ksh, HD, 16, 0);
#pragma unroll
    for (int kk = 0; kk < 4; ++kk) {
      v16h nb0 = kb0, nb1 = kb1;
      if (kk < 3) {
        nb0 = ld_frag(Ksh, HD, 0, (kk + 1) * 32);
        nb1 = ld_frag(Ksh, HD, 16, (kk + 1) * 32);
      }
      s0 = __builtin_amdgcn_wmma_f32_16x16x32_f16(false, qa[kk], false, kb0,
                                                  (short)0, s0, false, false);
      s1 = __builtin_amdgcn_wmma_f32_16x16x32_f16(false, qa[kk], false, kb1,
                                                  (short)0, s1, false, false);
      kb0 = nb0; kb1 = nb1;
    }
    // online softmax: row = vgpr + 8*lanehi, col = lane&15 (DPP16 rows)
#pragma unroll
    for (int v = 0; v < 8; ++v) {
      float cm = row16_max(fmaxf(s0[v], s1[v]));
      float nm = fmaxf(mrow[v], cm);
      float alpha = __expf(mrow[v] - nm);
      mrow[v] = nm;
      float p0 = __expf(s0[v] - nm);
      float p1 = __expf(s1[v] - nm);
      s0[v] = p0; s1[v] = p1;
      float rs = row16_sum(p0 + p1);
      lrow[v] = lrow[v] * alpha + rs;
#pragma unroll
      for (int d = 0; d < 4; ++d) oc[d][v] *= alpha;
    }
    // P (16x32) -> wave-private LDS -> A fragment
    _Float16* pw = PshAll + wave * 512;
#pragma unroll
    for (int v = 0; v < 8; ++v) {
      int row = v + 8 * lanehi;
      pw[row * 32 + (lane & 15)]      = (_Float16)s0[v];
      pw[row * 32 + 16 + (lane & 15)] = (_Float16)s1[v];
    }
    v16h pa = ld_frag(pw, 32, 0, 0);
    v16h bv = ld_frag(Vt, 32, 0, 0);
#pragma unroll
    for (int d = 0; d < 4; ++d) {
      v16h nbv = bv;
      if (d < 3) nbv = ld_frag(Vt, 32, (d + 1) * 16, 0);
      oc[d] = __builtin_amdgcn_wmma_f32_16x16x32_f16(false, pa, false, bv,
                                                     (short)0, oc[d], false, false);
      bv = nbv;
    }
    // ensure next chunk's DMA has landed before the whole WG flips buffers
    if (wave == 0)
      __builtin_amdgcn_s_wait_tensorcnt((unsigned short)0);
    __syncthreads();
  }
  // epilogue: o[b][s][h*64+d] = O / l
#pragma unroll
  for (int d = 0; d < 4; ++d)
#pragma unroll
    for (int v = 0; v < 8; ++v) {
      int row = qbase + v + 8 * lanehi;
      int ch = h * VD + d * 16 + (lane & 15);
      o_ws[((size_t)b * SDIM + row) * OCH + ch] = oc[d][v] / lrow[v];
    }
}

// ---------------------------------------------------------------------------
// K7: out conv (256 -> 128) + bias + residual.
// ---------------------------------------------------------------------------
__global__ __launch_bounds__(256) void outconv_kernel(
    const float* __restrict__ o_ws, const float* __restrict__ out_w,
    const float* __restrict__ out_b, const float* __restrict__ x,
    float* __restrict__ out) {
  __shared__ float osh[32][OCH];   // 32 KB
  const int tile = blockIdx.x;     // B*(S/32) = 256
  const int b = tile >> 7;
  const int s0 = (tile & 127) * 32;
  const int tid = threadIdx.x;
#pragma unroll
  for (int i = 0; i < (32 * OCH) / 256; ++i) {
    int idx = tid + i * 256;
    int sl = idx >> 8, cc = idx & 255;
    osh[sl][cc] = o_ws[((size_t)b * SDIM + s0 + sl) * OCH + cc];
  }
  __syncthreads();
  for (int i = 0; i < (DIMC * 32) / 256; ++i) {
    int item = tid + i * 256;
    int oc = item >> 5, sl = item & 31;
    float acc = out_b[oc];
    const float* wr = out_w + (size_t)oc * OCH;
    for (int c = 0; c < OCH; ++c) acc += wr[c] * osh[sl][c];
    int s = s0 + sl;
    out[((size_t)b * DIMC + oc) * SDIM + s] = acc + x[((size_t)b * DIMC + oc) * SDIM + s];
  }
}

// ---------------------------------------------------------------------------
extern "C" void kernel_launch(void* const* d_in, const int* in_sizes, int n_in,
                              void* d_out, int out_size, void* d_ws, size_t ws_size,
                              hipStream_t stream) {
  const float* x      = (const float*)d_in[0];
  const float* q1_w   = (const float*)d_in[1];
  const float* q1_b   = (const float*)d_in[2];
  const float* q_gn_g = (const float*)d_in[3];
  const float* q_gn_b = (const float*)d_in[4];
  const float* q2_w   = (const float*)d_in[5];
  const float* q2_b   = (const float*)d_in[6];
  const float* kv1_w  = (const float*)d_in[7];
  const float* kv1_b  = (const float*)d_in[8];
  const float* kv_gn_g= (const float*)d_in[9];
  const float* kv_gn_b= (const float*)d_in[10];
  const float* kvn_g  = (const float*)d_in[11];
  const float* kvn_b  = (const float*)d_in[12];
  const float* kvup_w = (const float*)d_in[13];
  const float* kvup_b = (const float*)d_in[14];
  const float* out_w  = (const float*)d_in[15];
  const float* out_b  = (const float*)d_in[16];
  float* out = (float*)d_out;

  char* ws = (char*)d_ws;
  size_t off = 0;
  auto carve = [&](size_t bytes) -> void* {
    off = (off + 255) & ~(size_t)255;
    void* p = ws + off;
    off += bytes;
    return p;
  };

  float*    q1_ws  = (float*)   carve((size_t)BATCH * QRANK * SDIM * 4);
  float*    kv1_ws = (float*)   carve((size_t)BATCH * DIMC * SDIM * 4);
  float*    kvc_ws = (float*)   carve((size_t)BATCH * KVRANK * SDIM * 4);
  _Float16* Qh     = (_Float16*)carve((size_t)BATCH * HEADS * SDIM * HD * 2);
  _Float16* Kh     = (_Float16*)carve((size_t)BATCH * HEADS * SDIM * HD * 2);
  _Float16* VhT    = (_Float16*)carve((size_t)BATCH * HEADS * SDIM * VD * 2);
  _Float16* Krope  = (_Float16*)carve((size_t)BATCH * SDIM * ROPE * 2);
  float*    o_ws   = (float*)   carve((size_t)BATCH * SDIM * OCH * 4);
  float*    statsB = (float*)   carve(256);
  float* stats1 = statsB;        // 8 floats
  float* stats2 = statsB + 16;   // 4 floats
  float* norm1  = statsB + 32;   // 8 floats
  float* norm2  = statsB + 48;   // 4 floats

  (void)hipMemsetAsync(statsB, 0, 256, stream);

  proj1_kernel<<<BATCH * (SDIM / 64), 256, 0, stream>>>(
      x, q1_w, q1_b, kv1_w, kv1_b, q1_ws, kv1_ws, stats1);
  stats1_kernel<<<1, 32, 0, stream>>>(stats1, norm1);
  mid_kernel<<<BATCH * (SDIM / 64), 256, 0, stream>>>(
      q1_ws, kv1_ws, norm1, q_gn_g, q_gn_b, q2_w, q2_b, kv_gn_g, kv_gn_b,
      Qh, Krope, kvc_ws, stats2);
  stats2_kernel<<<1, 32, 0, stream>>>(stats2, norm2);
  kvup_kernel<<<BATCH * (SDIM / 64), 256, 0, stream>>>(
      kvc_ws, norm2, kvn_g, kvn_b, kvup_w, kvup_b, Krope, Kh, VhT);
  attn_kernel<<<BATCH * HEADS * (SDIM / 64), 128, ATTN_LDS_BYTES, stream>>>(
      Qh, Kh, VhT, o_ws);
  outconv_kernel<<<BATCH * (SDIM / 32), 256, 0, stream>>>(o_ws, out_w, out_b, x, out);
}
